// NeighborhoodTransformerBlock_30777735643438
// MI455X (gfx1250) — compile-verified, matrix operand order checked by reading, not run on hardware
//
#include <hip/hip_runtime.h>
#include <hip/hip_bf16.h>
#include <math.h>

typedef _Float16 f16;
typedef __attribute__((ext_vector_type(16))) _Float16 v16h;
typedef __attribute__((ext_vector_type(8)))  float    v8f;
typedef __attribute__((ext_vector_type(4)))  unsigned u32x4;
typedef __attribute__((ext_vector_type(8)))  int      i32x8;
typedef __attribute__((ext_vector_type(4)))  int      i32x4;

#define TOKENS   65536       // 4 * 128 * 128
#define HW       16384       // 128*128
#define CDIM     96
#define HEADS    6
#define HD       16
#define KWIN     7
#define QKVC     288
#define HIDP     512         // padded 2*HID (510 -> 512)
#define HIDK     256         // padded HID (255 -> 256)

#define SMEM_ELEMS (32 * 256)

union FragU { v16h v; unsigned u[8]; };

#if __has_builtin(__builtin_amdgcn_tensor_load_to_lds)
#define HAVE_TDM 1
#endif

// ---------------------------------------------------------------------------
// Weight prep: convert f32 weights -> f16, with zero padding for pin/pout.
// ---------------------------------------------------------------------------
__global__ __launch_bounds__(256) void prep_weights(
    const float* __restrict__ qkv_w, const float* __restrict__ proj_w,
    const float* __restrict__ pin_w, const float* __restrict__ pout_w,
    f16* __restrict__ Wqkv, f16* __restrict__ Wproj,
    f16* __restrict__ Wpin, f16* __restrict__ Wpout)
{
    int i = blockIdx.x * 256 + threadIdx.x;            // 0 .. 110591
    if (i < 27648) {                                   // qkv [288][96]
        Wqkv[i] = (f16)qkv_w[i];
    } else if (i < 27648 + 9216) {                     // proj [96][96]
        int j = i - 27648;
        Wproj[j] = (f16)proj_w[j];
    } else if (i < 27648 + 9216 + 49152) {             // pin [512][96], rows 510..511 zero
        int j = i - (27648 + 9216);
        int r = j / 96, c = j % 96;
        Wpin[j] = (f16)(r < 510 ? pin_w[r * 96 + c] : 0.f);
    } else {                                           // pout [96][256], cols 255 zero
        int j = i - (27648 + 9216 + 49152);
        int r = j / 256, c = j % 256;
        Wpout[j] = (f16)(c < 255 ? pout_w[r * 255 + c] : 0.f);
    }
}

// ---------------------------------------------------------------------------
// LayerNorm over channels (wave-per-pixel, wave32 shfl reductions).
// ---------------------------------------------------------------------------
__global__ __launch_bounds__(256) void layernorm_k(
    const float* __restrict__ in, int nchw,
    const float* __restrict__ w, const float* __restrict__ b,
    f16* __restrict__ outH, float* __restrict__ residOut)
{
    int lane = threadIdx.x & 31;
    int wv   = threadIdx.x >> 5;
    int t    = blockIdx.x * 8 + wv;                    // pixel id, < TOKENS
    int bi   = t >> 14;                                // batch
    int hw   = t & (HW - 1);

    float vals[3];
    float s = 0.f, sq = 0.f;
    #pragma unroll
    for (int j = 0; j < 3; ++j) {
        int c = lane + 32 * j;
        long idx = nchw ? ((long)(bi * CDIM + c)) * HW + hw
                        : (long)t * CDIM + c;
        float v = in[idx];
        vals[j] = v;
        s += v; sq += v * v;
    }
    #pragma unroll
    for (int off = 16; off > 0; off >>= 1) {
        s  += __shfl_xor(s,  off, 32);
        sq += __shfl_xor(sq, off, 32);
    }
    const float inv96 = 1.0f / 96.0f;
    float mean = s * inv96;
    float var  = sq * inv96 - mean * mean;             // biased, matches reference
    float rstd = rsqrtf(var + 1e-5f);
    #pragma unroll
    for (int j = 0; j < 3; ++j) {
        int c = lane + 32 * j;
        float y = (vals[j] - mean) * rstd * w[c] + b[c];
        outH[(long)t * CDIM + c] = (f16)y;
        if (residOut) residOut[(long)t * CDIM + c] = vals[j];
    }
}

// ---------------------------------------------------------------------------
// Wave-tiled GEMM: D[m][n] = sum_k A[m][k] * B[n][k]   (B stored [N][K])
// Block = 8 waves, block tile = 256(M) x 32(N). Each wave owns 32x32
// (2x2 of 16x16 WMMA tiles). The block's 32xK B-panel is staged into LDS
// once via the Tensor Data Mover (tensor_load_to_lds, TENSORcnt), then
// B fragments come from LDS (ds_load_b128) while A streams from global.
// ---------------------------------------------------------------------------
__global__ __launch_bounds__(256) void gemm16_wmma(
    const f16* __restrict__ A, int lda,
    const f16* __restrict__ B, int K, int N, int nblocks,
    const float* __restrict__ bias,
    const float* __restrict__ scale,
    f16* __restrict__ outH, int ldo,
    float* __restrict__ resid,
    float* __restrict__ outNCHW,
    int mode)
{
    __shared__ f16 smemB[SMEM_ELEMS + 16];             // 16 KB panel + pad slot

    int lane = threadIdx.x & 31;
    int wave = threadIdx.x >> 5;
    int bm = blockIdx.x / nblocks;                     // 256-row super-tile
    int bn = blockIdx.x % nblocks;                     // 32-col panel

    // ---- stage B panel [bn*32 .. bn*32+32) x K into LDS via TDM ----
#ifdef HAVE_TDM
    if (threadIdx.x < 32) {                            // wave 0 issues the DMA
        unsigned long long gaddr =
            (unsigned long long)(const void*)(B + (size_t)bn * 32 * K);
        u32x4 g0;
        g0[0] = 1u;                                    // count=1, no gather
        g0[1] = 0u;                                    // lds_addr = 0 (smemB base)
        g0[2] = (unsigned)(gaddr & 0xFFFFFFFFu);
        g0[3] = (unsigned)((gaddr >> 32) & 0x01FFFFFFu) | 0x80000000u; // type=2
        i32x8 g1;
        unsigned uK = (unsigned)K, uN = (unsigned)N;
        g1[0] = 0x00010000;                            // wg_mask=0, data_size=2B
        g1[1] = (int)((uK & 0xFFFFu) << 16);           // tensor_dim0[15:0]
        g1[2] = (int)((uK >> 16) | ((uN & 0xFFFFu) << 16));   // dim0 hi | dim1 lo
        g1[3] = (int)((uN >> 16) | ((uK & 0xFFFFu) << 16));   // dim1 hi | tile_dim0=K
        g1[4] = 32;                                    // tile_dim1 = 32 rows
        g1[5] = (int)uK;                               // tensor_dim0_stride lo32 = K
        g1[6] = (int)((uK & 0xFFFFu) << 16);           // stride0 hi=0 | stride1 lo = K
        g1[7] = (int)(uK >> 16);                       // stride1 hi
        i32x4 gz = {0, 0, 0, 0};
#if defined(__clang_major__) && (__clang_major__ >= 23)
        i32x8 gz8 = {0, 0, 0, 0, 0, 0, 0, 0};
        __builtin_amdgcn_tensor_load_to_lds(g0, g1, gz, gz, gz8, 0);
#else
        __builtin_amdgcn_tensor_load_to_lds(g0, g1, gz, gz, 0);
#endif
#if __has_builtin(__builtin_amdgcn_s_wait_tensorcnt)
        __builtin_amdgcn_s_wait_tensorcnt(0);
#else
        asm volatile("s_wait_tensorcnt 0x0" ::: "memory");
#endif
    }
    // Make the TDM's LDS write visible to alias analysis WITHOUT an "m"
    // constraint (backend can't select LDS inline-asm memory operands):
    // (a) escape the pointer into an asm with a memory clobber, and
    // (b) one volatile store to a pad slot so the array has a real store.
    asm volatile("" : : "v"((const void*)smemB) : "memory");
    if (threadIdx.x == 0) ((volatile f16*)smemB)[SMEM_ELEMS] = (f16)0;
#else
    {   // fallback: cooperative load of the B panel
        int total = 32 * K;                            // f16 elements
        const f16* src = B + (size_t)bn * 32 * K;
        for (int i = threadIdx.x; i < total; i += 256) smemB[i] = src[i];
    }
#endif
    __syncthreads();

    // ---- compute: wave covers rows [row_base, row_base+32), cols [bn*32, +32) ----
    int row_base = bm * 256 + wave * 32;
    int mr  = lane & 15;
    int half = lane >> 4;

    const f16* ap0 = A + (long)(row_base + mr)      * lda;   // m-tile 0
    const f16* ap1 = A + (long)(row_base + 16 + mr) * lda;   // m-tile 1
    const f16* bp0 = smemB + (long)mr        * K;            // n-tile 0 (LDS)
    const f16* bp1 = smemB + (long)(16 + mr) * K;            // n-tile 1 (LDS)

    v8f acc00 = {}, acc01 = {}, acc10 = {}, acc11 = {};
    for (int k0 = 0; k0 < K; k0 += 32) {
        if (k0 + 32 < K) {                             // near-cache prefetch of A
            __builtin_prefetch(ap0 + k0 + 32, 0, 3);
            __builtin_prefetch(ap1 + k0 + 32, 0, 3);
        }
        FragU a0, a1, b0, b1;
        #pragma unroll
        for (int j = 0; j < 4; ++j) {
            int klo = k0 + 8 * half + 2 * j;
            int khi = klo + 16;
            a0.u[j]     = *(const unsigned*)(ap0 + klo);
            a0.u[4 + j] = *(const unsigned*)(ap0 + khi);
            a1.u[j]     = *(const unsigned*)(ap1 + klo);
            a1.u[4 + j] = *(const unsigned*)(ap1 + khi);
            b0.u[j]     = *(const unsigned*)(bp0 + klo);
            b0.u[4 + j] = *(const unsigned*)(bp0 + khi);
            b1.u[j]     = *(const unsigned*)(bp1 + klo);
            b1.u[4 + j] = *(const unsigned*)(bp1 + khi);
        }
        acc00 = __builtin_amdgcn_wmma_f32_16x16x32_f16(false, a0.v, false, b0.v, (short)0, acc00, false, false);
        acc01 = __builtin_amdgcn_wmma_f32_16x16x32_f16(false, a0.v, false, b1.v, (short)0, acc01, false, false);
        acc10 = __builtin_amdgcn_wmma_f32_16x16x32_f16(false, a1.v, false, b0.v, (short)0, acc10, false, false);
        acc11 = __builtin_amdgcn_wmma_f32_16x16x32_f16(false, a1.v, false, b1.v, (short)0, acc11, false, false);
    }

    v8f accs[2][2] = {{acc00, acc01}, {acc10, acc11}};
    #pragma unroll
    for (int ti = 0; ti < 2; ++ti) {
        #pragma unroll
        for (int tj = 0; tj < 2; ++tj) {
            int col = bn * 32 + tj * 16 + (lane & 15);
            float bv = bias  ? bias[col]  : 0.f;
            float sv = scale ? scale[col] : 0.f;
            #pragma unroll
            for (int i = 0; i < 8; ++i) {
                long row = (long)row_base + ti * 16 + i + 8 * half;
                float v = accs[ti][tj][i];
                if (mode == 0) {
                    outH[row * ldo + col] = (f16)(v + bv);
                } else if (mode == 1) {
                    long ix = row * CDIM + col;
                    resid[ix] = resid[ix] + sv * (v + bv);
                } else if (mode == 2) {
                    outH[row * ldo + col] = (f16)v;
                } else {
                    long ix = row * CDIM + col;
                    float y = resid[ix] + sv * v;
                    int bb = (int)(row >> 14);
                    int hw = (int)(row & (HW - 1));
                    outNCHW[((long)(bb * CDIM + col)) * HW + hw] = y;
                }
            }
        }
    }
}

// ---------------------------------------------------------------------------
// Depthwise 3x3 'SAME' conv, NHWC, f16 input. One thread per (pixel, channel).
// ---------------------------------------------------------------------------
__global__ __launch_bounds__(256) void dwconv3x3_k(
    const f16* __restrict__ in, int ldin, int C,
    const float* __restrict__ w9, const float* __restrict__ bias,
    float* __restrict__ outF, f16* __restrict__ outH, int ldout)
{
    long idx = (long)blockIdx.x * 256 + threadIdx.x;
    long total = (long)TOKENS * C;
    if (idx >= total) return;
    int c = (int)(idx % C);
    int t = (int)(idx / C);
    int bI = t >> 14;
    int hw = t & (HW - 1);
    int h  = hw >> 7, wcol = hw & 127;

    float acc = bias ? bias[c] : 0.f;
    #pragma unroll
    for (int dh = 0; dh < 3; ++dh) {
        int hh = h + dh - 1;
        if (hh < 0 || hh > 127) continue;
        #pragma unroll
        for (int dw = 0; dw < 3; ++dw) {
            int ww = wcol + dw - 1;
            if (ww < 0 || ww > 127) continue;
            long src = ((long)(bI * HW + hh * 128 + ww)) * ldin + c;
            acc += (float)in[src] * w9[c * 9 + dh * 3 + dw];
        }
    }
    if (outF) outF[(long)t * ldout + c] = acc;
    else      outH[(long)t * ldout + c] = (f16)acc;
}

// ---------------------------------------------------------------------------
// L2-normalize q and k per (pixel, head) in qkv buffer [t][288] f32, in place.
// ---------------------------------------------------------------------------
__global__ __launch_bounds__(256) void l2norm_qk(float* __restrict__ qkv)
{
    int idx = blockIdx.x * 256 + threadIdx.x;         // TOKENS * 12
    int r   = idx % 12;
    int t   = idx / 12;
    int head = r >> 1;
    int isK  = r & 1;
    float* p = qkv + (long)t * QKVC + isK * CDIM + head * HD;
    float s = 0.f;
    #pragma unroll
    for (int d = 0; d < HD; ++d) s += p[d] * p[d];
    float scl = 1.0f / fmaxf(sqrtf(s), 1e-12f);
    #pragma unroll
    for (int d = 0; d < HD; ++d) p[d] *= scl;
}

// ---------------------------------------------------------------------------
// Neighborhood attention: 7x7 edge-clamped window, RPB, temperature, softmax.
// One thread per (pixel, head). Output f16 [t][96].
// ---------------------------------------------------------------------------
__global__ __launch_bounds__(256) void natten_k(
    const float* __restrict__ qkv, const float* __restrict__ rpb,
    const float* __restrict__ temp, f16* __restrict__ outH)
{
    int idx  = blockIdx.x * 256 + threadIdx.x;        // TOKENS * HEADS
    int head = idx % HEADS;
    int t    = idx / HEADS;
    int bI = t >> 14;
    int hw = t & (HW - 1);
    int h  = hw >> 7, w = hw & 127;
    int r0 = min(max(h - 3, 0), 121);
    int c0 = min(max(w - 3, 0), 121);

    float q[HD];
    const float* qp = qkv + (long)t * QKVC + head * HD;
    #pragma unroll
    for (int d = 0; d < HD; ++d) q[d] = qp[d];

    const float* rpbh = rpb + head * 13 * 13;
    float T = temp[head];

    float s[49];
    float mx = -3.0e38f;
    int o = 0;
    for (int ki = 0; ki < KWIN; ++ki) {
        int r  = r0 + ki;
        int bi = h - r + 6;
        for (int kj = 0; kj < KWIN; ++kj, ++o) {
            int c  = c0 + kj;
            int bj = w - c + 6;
            const float* kp = qkv + ((long)(bI * HW + r * 128 + c)) * QKVC
                              + CDIM + head * HD;
            float dot = 0.f;
            #pragma unroll
            for (int d = 0; d < HD; ++d) dot += q[d] * kp[d];
            float sc = (dot + rpbh[bi * 13 + bj]) * T;
            s[o] = sc;
            mx = fmaxf(mx, sc);
        }
    }
    float den = 0.f;
    #pragma unroll
    for (o = 0; o < 49; ++o) { s[o] = __expf(s[o] - mx); den += s[o]; }
    float inv = 1.0f / den;

    float outv[HD];
    #pragma unroll
    for (int d = 0; d < HD; ++d) outv[d] = 0.f;
    o = 0;
    for (int ki = 0; ki < KWIN; ++ki) {
        int r = r0 + ki;
        for (int kj = 0; kj < KWIN; ++kj, ++o) {
            int c = c0 + kj;
            const float* vp = qkv + ((long)(bI * HW + r * 128 + c)) * QKVC
                              + 2 * CDIM + head * HD;
            float a = s[o] * inv;
            #pragma unroll
            for (int d = 0; d < HD; ++d) outv[d] += a * vp[d];
        }
    }
    #pragma unroll
    for (int d = 0; d < HD; ++d)
        outH[(long)t * CDIM + head * HD + d] = (f16)outv[d];
}

// ---------------------------------------------------------------------------
// GELU gate: gated[t][j] = gelu(y[t][j]) * y[t][255+j], j<255 (col 255 -> 0).
// ---------------------------------------------------------------------------
__global__ __launch_bounds__(256) void gelu_gate_k(
    const f16* __restrict__ y, f16* __restrict__ gated)
{
    int idx = blockIdx.x * 256 + threadIdx.x;         // TOKENS * 256
    int j = idx & 255;
    int t = idx >> 8;
    float out = 0.f;
    if (j < 255) {
        float a = (float)y[(long)t * HIDP + j];
        float b = (float)y[(long)t * HIDP + 255 + j];
        float g = 0.5f * a * (1.0f + erff(a * 0.70710678118654752f));
        out = g * b;
    }
    gated[(long)t * HIDK + j] = (f16)out;
}

// ---------------------------------------------------------------------------
// Host launcher
// ---------------------------------------------------------------------------
extern "C" void kernel_launch(void* const* d_in, const int* in_sizes, int n_in,
                              void* d_out, int out_size, void* d_ws, size_t ws_size,
                              hipStream_t stream) {
    (void)in_sizes; (void)n_in; (void)out_size; (void)ws_size;

    const float* x        = (const float*)d_in[0];
    const float* ln1_w    = (const float*)d_in[1];
    const float* ln1_b    = (const float*)d_in[2];
    const float* qkv_w    = (const float*)d_in[3];
    const float* qkv_b    = (const float*)d_in[4];
    const float* qkv_dw_w = (const float*)d_in[5];
    const float* qkv_dw_b = (const float*)d_in[6];
    const float* rpb      = (const float*)d_in[7];
    const float* temp     = (const float*)d_in[8];
    const float* proj_w   = (const float*)d_in[9];
    const float* proj_b   = (const float*)d_in[10];
    const float* ln2_w    = (const float*)d_in[11];
    const float* ln2_b    = (const float*)d_in[12];
    const float* pin_w    = (const float*)d_in[13];
    const float* dw_w     = (const float*)d_in[14];
    const float* pout_w   = (const float*)d_in[15];
    const float* beta     = (const float*)d_in[16];
    const float* gamma    = (const float*)d_in[17];
    float* out            = (float*)d_out;

    // ---- workspace carve-up (256B aligned) ----
    char* ws = (char*)d_ws;
    size_t off = 0;
    auto carve = [&](size_t bytes) {
        size_t o = off;
        off = (off + bytes + 255) & ~(size_t)255;
        return ws + o;
    };
    float* resid = (float*)carve((size_t)TOKENS * CDIM * 4);   // NHWC f32 running x
    f16*   hbuf  = (f16*)  carve((size_t)TOKENS * CDIM * 2);   // ln1 / attn-out / ln2
    f16*   bufA  = (f16*)  carve((size_t)TOKENS * QKVC * 2);   // qkv pre-dw / gated
    float* bufB  = (float*)carve((size_t)TOKENS * QKVC * 4);   // qkv post-dw / pin f16
    f16*   bufC  = (f16*)  carve((size_t)TOKENS * HIDP * 2);   // ffn post-dw
    f16*   Wqkv  = (f16*)  carve(27648 * 2);
    f16*   Wproj = (f16*)  carve(9216 * 2);
    f16*   Wpin  = (f16*)  carve(49152 * 2);
    f16*   Wpout = (f16*)  carve(24576 * 2);
    f16*   pinH  = (f16*)bufB;                                 // reuse bufB as f16 [t][512]

    const int MB = TOKENS / 256;                               // 256-row super-tiles

    // 1. weight prep (110592 elems)
    prep_weights<<<432, 256, 0, stream>>>(qkv_w, proj_w, pin_w, pout_w,
                                          Wqkv, Wproj, Wpin, Wpout);
    // 2. LN1 (NCHW read) -> hbuf f16, resid f32
    layernorm_k<<<TOKENS / 8, 256, 0, stream>>>(x, 1, ln1_w, ln1_b, hbuf, resid);
    // 3. qkv 1x1: [65536,96] x [288,96]^T -> bufA f16, +bias     (WMMA + TDM)
    gemm16_wmma<<<MB * (QKVC / 32), 256, 0, stream>>>(
        hbuf, CDIM, Wqkv, CDIM, QKVC, QKVC / 32,
        qkv_b, nullptr, bufA, QKVC, nullptr, nullptr, 0);
    // 4. depthwise 3x3 on 288 ch (+bias) -> bufB f32
    dwconv3x3_k<<<(TOKENS * QKVC) / 256, 256, 0, stream>>>(
        bufA, QKVC, QKVC, qkv_dw_w, qkv_dw_b, bufB, nullptr, QKVC);
    // 5. l2-normalize q,k per (pixel, head)
    l2norm_qk<<<(TOKENS * 12) / 256, 256, 0, stream>>>(bufB);
    // 6. neighborhood attention -> hbuf f16
    natten_k<<<(TOKENS * HEADS) / 256, 256, 0, stream>>>(bufB, rpb, temp, hbuf);
    // 7. proj 1x1 + residual: resid += beta*(attn@Wproj + b)     (WMMA + TDM)
    gemm16_wmma<<<MB * (CDIM / 32), 256, 0, stream>>>(
        hbuf, CDIM, Wproj, CDIM, CDIM, CDIM / 32,
        proj_b, beta, nullptr, 0, resid, nullptr, 1);
    // 8. LN2 (NHWC resid read) -> hbuf f16
    layernorm_k<<<TOKENS / 8, 256, 0, stream>>>(resid, 0, ln2_w, ln2_b, hbuf, nullptr);
    // 9. pin 1x1: [65536,96] x [512,96]^T -> pinH f16 (no bias)  (WMMA + TDM)
    gemm16_wmma<<<MB * (HIDP / 32), 256, 0, stream>>>(
        hbuf, CDIM, Wpin, CDIM, HIDP, HIDP / 32,
        nullptr, nullptr, pinH, HIDP, nullptr, nullptr, 2);
    // 10. depthwise 3x3 on 510 ch (no bias) -> bufC f16
    dwconv3x3_k<<<(TOKENS * 510 + 255) / 256, 256, 0, stream>>>(
        pinH, HIDP, 510, dw_w, nullptr, nullptr, bufC, HIDP);
    // 11. gelu gate -> bufA f16 [t][256] (K-padded)
    gelu_gate_k<<<(TOKENS * HIDK) / 256, 256, 0, stream>>>(bufC, bufA);
    // 12. pout 1x1 + residual -> d_out NCHW f32                  (WMMA + TDM)
    gemm16_wmma<<<MB * (CDIM / 32), 256, 0, stream>>>(
        bufA, HIDK, Wpout, HIDK, CDIM, CDIM / 32,
        nullptr, gamma, nullptr, 0, resid, out, 3);
}
